// TransformerAutoRNet_56796647522375
// MI455X (gfx1250) — compile-verified, matrix operand order checked by reading, not run on hardware
//
#include <hip/hip_runtime.h>
#include <hip/hip_bf16.h>
#include <math.h>

typedef __attribute__((ext_vector_type(16))) _Float16 v16h;
typedef __attribute__((ext_vector_type(8)))  _Float16 v8h;
typedef __attribute__((ext_vector_type(8)))  float    v8f;

#define BQ    2
#define TCQ   256
#define TTQ   256
#define NSEQ  512          // TC+TT per batch
#define ROWS  1024         // B*NSEQ
#define DMQ   512
#define HQ    8
#define DQ    64
#define DFFQ  256
#define OUTQ  8
#define XCAT  24           // X_DIM + Y_DIM

// ---------------------------------------------------------------------------
// Build concatenated input (B, N, 24): rows 0..255 = [cx|cy], 256..511 = [tx|-99.9]
// ---------------------------------------------------------------------------
__global__ void build_input_k(const float* __restrict__ cx, const float* __restrict__ cy,
                              const float* __restrict__ tx, float* __restrict__ Xin) {
    int idx = blockIdx.x * blockDim.x + threadIdx.x;
    if (idx >= ROWS * XCAT) return;
    int row = idx / XCAT, c = idx % XCAT;
    int b = row / NSEQ, n = row % NSEQ;
    float v;
    if (n < TCQ) {
        v = (c < 16) ? cx[(b * TCQ + n) * 16 + c] : cy[(b * TCQ + n) * 8 + (c - 16)];
    } else {
        int t = n - TCQ;
        v = (c < 16) ? tx[(b * TTQ + t) * 16 + c] : -99.9f;
    }
    Xin[idx] = v;
}

// ---------------------------------------------------------------------------
// BatchNorm stats over all (b,n): stats[c]=mu, stats[24+c]=rsqrt(var+eps)
// ---------------------------------------------------------------------------
__global__ void bn_stats_k(const float* __restrict__ Xin, float* __restrict__ stats) {
    int c = threadIdx.x;
    if (c >= XCAT) return;
    float s = 0.f, ss = 0.f;
    for (int r = 0; r < ROWS; ++r) {
        float v = Xin[r * XCAT + c];
        s += v; ss += v * v;
    }
    float mu = s / (float)ROWS;
    float var = ss / (float)ROWS - mu * mu;
    stats[c]        = mu;
    stats[XCAT + c] = rsqrtf(var + 1e-5f);
}

// ---------------------------------------------------------------------------
// Encode: X = ((Xin - mu) * rstd) @ enc_w + enc_b  -> f32 X and f16 Xh
// ---------------------------------------------------------------------------
__global__ void encode_k(const float* __restrict__ Xin, const float* __restrict__ stats,
                         const float* __restrict__ ew, const float* __restrict__ eb,
                         float* __restrict__ X, _Float16* __restrict__ Xh) {
    int idx = blockIdx.x * blockDim.x + threadIdx.x;
    if (idx >= ROWS * DMQ) return;
    int row = idx / DMQ, j = idx % DMQ;
    float acc = eb[j];
    #pragma unroll
    for (int c = 0; c < XCAT; ++c) {
        float xn = (Xin[row * XCAT + c] - stats[c]) * stats[XCAT + c];
        acc += xn * ew[c * DMQ + j];
    }
    X[idx]  = acc;
    Xh[idx] = (_Float16)acc;
}

// ---------------------------------------------------------------------------
// Transpose + cast weight: W (K x N) f32 -> Wt (N x K) f16
// ---------------------------------------------------------------------------
__global__ void transpose_cast_k(const float* __restrict__ W, _Float16* __restrict__ Wt,
                                 int K, int N) {
    int idx = blockIdx.x * blockDim.x + threadIdx.x;
    if (idx >= K * N) return;
    int k = idx / N, n = idx % N;
    Wt[(size_t)n * K + k] = (_Float16)W[idx];
}

// ---------------------------------------------------------------------------
// WMMA GEMM, register-blocked: each wave computes a 32(M) x 64(N) macro-tile
// (2x4 = 8 accumulator tiles). A frag reused across 4 N-tiles, B frag across
// 2 M-tiles -> 12 b128 loads per 8 v_wmma_f32_16x16x32_f16.
// mode: 0 = none, 1 = elu(x)+1, 2 = exact gelu
// Output: f32 to C if Ch==nullptr, else f16 to Ch.
// ---------------------------------------------------------------------------
__global__ void gemm_wmma_k(const _Float16* __restrict__ A, const _Float16* __restrict__ Wt,
                            const float* __restrict__ bias, float* __restrict__ C,
                            _Float16* __restrict__ Ch,
                            int M, int N, int K, int mode) {
    int wave = threadIdx.x >> 5;
    int lane = threadIdx.x & 31;
    int nt = N >> 6;                            // macro-tiles along N (64 cols)
    int mt = M >> 5;                            // macro-tiles along M (32 rows)
    int tile = blockIdx.x * (blockDim.x >> 5) + wave;
    if (tile >= mt * nt) return;                // uniform per wave, EXEC stays all-1
    int tn = tile % nt;
    int tm = tile / nt;
    int lm   = lane & 15;
    int koff = (lane >> 4) << 3;                // 0 for lanes 0-15, 8 for lanes 16-31

    const _Float16* arow0 = A + (size_t)(tm * 32 + lm) * K + koff;
    const _Float16* arow1 = arow0 + (size_t)16 * K;
    const _Float16* brow[4];
    #pragma unroll
    for (int j = 0; j < 4; ++j)
        brow[j] = Wt + (size_t)(tn * 64 + j * 16 + lm) * K + koff;

    v8f acc[2][4];
    #pragma unroll
    for (int j = 0; j < 4; ++j) {
        float bv = bias[tn * 64 + j * 16 + lm];
        #pragma unroll
        for (int r = 0; r < 8; ++r) { acc[0][j][r] = bv; acc[1][j][r] = bv; }
    }

    for (int kk = 0; kk < K; kk += 32) {
        if (kk + 64 < K) {                       // gfx1250 global_prefetch path
            __builtin_prefetch(arow0 + kk + 64, 0, 0);
            __builtin_prefetch(brow[0] + kk + 64, 0, 0);
        }
        union { v16h v; v8h h[2]; } a0, a1;
        a0.h[0] = *(const v8h*)(arow0 + kk);        // K = base .. base+7
        a0.h[1] = *(const v8h*)(arow0 + kk + 16);   // K = base+16 .. base+23
        a1.h[0] = *(const v8h*)(arow1 + kk);
        a1.h[1] = *(const v8h*)(arow1 + kk + 16);
        #pragma unroll
        for (int j = 0; j < 4; ++j) {
            union { v16h v; v8h h[2]; } bb;
            bb.h[0] = *(const v8h*)(brow[j] + kk);
            bb.h[1] = *(const v8h*)(brow[j] + kk + 16);
            acc[0][j] = __builtin_amdgcn_wmma_f32_16x16x32_f16(
                            false, a0.v, false, bb.v, (short)0, acc[0][j], false, false);
            acc[1][j] = __builtin_amdgcn_wmma_f32_16x16x32_f16(
                            false, a1.v, false, bb.v, (short)0, acc[1][j], false, false);
        }
    }

    #pragma unroll
    for (int i = 0; i < 2; ++i) {
        int rowbase = tm * 32 + i * 16 + koff;      // +8 for upper half-lanes
        #pragma unroll
        for (int j = 0; j < 4; ++j) {
            int col = tn * 64 + j * 16 + lm;
            #pragma unroll
            for (int r = 0; r < 8; ++r) {
                float v = acc[i][j][r];
                if (mode == 1)      v = (v > 0.f) ? (v + 1.f) : expf(v);  // elu+1
                else if (mode == 2) v = 0.5f * v * (1.f + erff(v * 0.70710678118654752f));
                size_t oidx = (size_t)(rowbase + r) * N + col;
                if (Ch) Ch[oidx] = (_Float16)v;
                else    C[oidx]  = v;
            }
        }
    }
}

// ---------------------------------------------------------------------------
// Causal linear attention recurrence, one block per (b,h). kv state in LDS.
// Q,K already have elu+1 applied (fused in GEMM). Output (f16) = num/(den+eps).
// ---------------------------------------------------------------------------
__global__ void attn_k(const float* __restrict__ Q, const float* __restrict__ K,
                       const float* __restrict__ V, _Float16* __restrict__ AOh) {
    __shared__ float kv[DQ * DQ];
    __shared__ float kvec[DQ], vvec[DQ], qvec[DQ], ksum[DQ];
    __shared__ float np[4 * DQ], pd[DQ];
    int b = blockIdx.x >> 3;
    int h = blockIdx.x & 7;
    int t = threadIdx.x;                         // 256 threads
    for (int i = t; i < DQ * DQ; i += 256) kv[i] = 0.f;
    if (t < DQ) ksum[t] = 0.f;
    __syncthreads();

    for (int n = 0; n < NSEQ; ++n) {
        size_t base = ((size_t)(b * NSEQ + n)) * DMQ + h * DQ;
        if (t < DQ) { kvec[t] = K[base + t]; vvec[t] = V[base + t]; qvec[t] = Q[base + t]; }
        __syncthreads();

        int i0 = t * 16;                         // 16 kv entries per thread
        #pragma unroll
        for (int i = 0; i < 16; ++i) {
            int idx = i0 + i;
            kv[idx] += kvec[idx >> 6] * vvec[idx & 63];
        }
        if (t < DQ) ksum[t] += kvec[t];
        __syncthreads();

        int e = t & 63, part = t >> 6;
        float s = 0.f;
        #pragma unroll
        for (int i = 0; i < 16; ++i) {
            int d = part * 16 + i;
            s += qvec[d] * kv[d * DQ + e];
        }
        np[part * DQ + e] = s;
        if (t < DQ) pd[t] = qvec[t] * ksum[t];
        __syncthreads();

        if (t < DQ) {
            float den = 0.f;
            #pragma unroll
            for (int d = 0; d < DQ; ++d) den += pd[d];
            float num = np[t] + np[DQ + t] + np[2 * DQ + t] + np[3 * DQ + t];
            AOh[base + t] = (_Float16)(num / (den + 1e-6f));
        }
        __syncthreads();
    }
}

// ---------------------------------------------------------------------------
// LayerNorm with residual: X = LN(X + R) * s + b, writes f32 X and f16 Xh
// One block (128 threads x 4 elems) per row.
// ---------------------------------------------------------------------------
__global__ void ln_res_k(float* __restrict__ X, _Float16* __restrict__ Xh,
                         const float* __restrict__ R,
                         const float* __restrict__ gs, const float* __restrict__ gb) {
    __shared__ float rs[128], rss[128];
    int row = blockIdx.x, t = threadIdx.x;
    size_t base = (size_t)row * DMQ;
    float v[4];
    float s = 0.f, ss = 0.f;
    #pragma unroll
    for (int i = 0; i < 4; ++i) {
        int j = t * 4 + i;
        v[i] = X[base + j] + R[base + j];
        s += v[i]; ss += v[i] * v[i];
    }
    rs[t] = s; rss[t] = ss;
    __syncthreads();
    for (int k = 64; k > 0; k >>= 1) {
        if (t < k) { rs[t] += rs[t + k]; rss[t] += rss[t + k]; }
        __syncthreads();
    }
    float mean = rs[0] / (float)DMQ;
    float var  = rss[0] / (float)DMQ - mean * mean;
    float inv  = rsqrtf(var + 1e-5f);
    #pragma unroll
    for (int i = 0; i < 4; ++i) {
        int j = t * 4 + i;
        float o = (v[i] - mean) * inv * gs[j] + gb[j];
        X[base + j]  = o;
        Xh[base + j] = (_Float16)o;
    }
}

// ---------------------------------------------------------------------------
// Head: mean/log_sigma (512->8 per target token), per-elem log-prob & sq-err
// ---------------------------------------------------------------------------
__global__ void head_k(const float* __restrict__ X, const float* __restrict__ ty,
                       const float* __restrict__ mw, const float* __restrict__ mb,
                       const float* __restrict__ sw, const float* __restrict__ sb,
                       float* __restrict__ out, float* __restrict__ lparr,
                       float* __restrict__ sqarr) {
    int idx = blockIdx.x * blockDim.x + threadIdx.x;  // B*TT*OUT = 4096
    if (idx >= BQ * TTQ * OUTQ) return;
    int b = idx / (TTQ * OUTQ);
    int rem = idx % (TTQ * OUTQ);
    int tp = rem / OUTQ, o = rem % OUTQ;
    size_t row = (size_t)(b * NSEQ + TCQ + tp);
    const float* xr = X + row * DMQ;
    float m = mb[o], ls = sb[o];
    for (int k = 0; k < DMQ; ++k) {
        float xv = xr[k];
        m  += xv * mw[k * OUTQ + o];
        ls += xv * sw[k * OUTQ + o];
    }
    m  = fmaxf(m, 0.f);
    ls = fmaxf(ls, 0.f);
    float sp    = (ls > 20.f) ? ls : log1pf(expf(ls));
    float sigma = 0.01f + 0.99f * sp;
    float tyv   = ty[(b * TTQ + tp) * OUTQ + o];
    float z     = (tyv - m) / sigma;
    float lp    = -0.5f * z * z - logf(sigma) - 0.91893853320467274f;
    out[idx]                          = m;    // y_pred
    out[BQ * TTQ * OUTQ + 3 + idx]    = ls;   // log_sigma after 3 scalar losses
    lparr[idx] = lp;
    sqarr[idx] = (m - tyv) * (m - tyv);
}

// ---------------------------------------------------------------------------
// Final loss reductions (single block)
// ---------------------------------------------------------------------------
__global__ void loss_k(const float* __restrict__ lparr, const float* __restrict__ sqarr,
                       float* __restrict__ out) {
    __shared__ float r1[256], r2[256], r3[256];
    int t = threadIdx.x;
    float s1 = 0.f, s2 = 0.f, s3 = 0.f;
    for (int p = t; p < BQ * TTQ; p += 256) {
        int tp = p % TTQ;
        float sum = 0.f;
        #pragma unroll
        for (int o = 0; o < OUTQ; ++o) sum += lparr[p * OUTQ + o];
        float lossp = -sum / (float)OUTQ;
        s1 += lossp;
        s2 += lossp * rsqrtf((float)(tp + 1));
    }
    for (int i = t; i < BQ * TTQ * OUTQ; i += 256) s3 += sqarr[i];
    r1[t] = s1; r2[t] = s2; r3[t] = s3;
    __syncthreads();
    for (int k = 128; k > 0; k >>= 1) {
        if (t < k) { r1[t] += r1[t + k]; r2[t] += r2[t + k]; r3[t] += r3[t + k]; }
        __syncthreads();
    }
    if (t == 0) {
        int np = BQ * TTQ;
        out[BQ * TTQ * OUTQ + 0] = r1[0] / (float)np;                  // loss_p mean
        out[BQ * TTQ * OUTQ + 1] = r3[0] / (float)(BQ * TTQ * OUTQ);   // loss_mse
        out[BQ * TTQ * OUTQ + 2] = r2[0] / (float)np;                  // loss_p weighted
    }
}

// ---------------------------------------------------------------------------
// Host-side orchestration
// ---------------------------------------------------------------------------
extern "C" void kernel_launch(void* const* d_in, const int* in_sizes, int n_in,
                              void* d_out, int out_size, void* d_ws, size_t ws_size,
                              hipStream_t stream) {
    (void)in_sizes; (void)n_in; (void)out_size; (void)ws_size;
    const float* cx  = (const float*)d_in[0];
    const float* cy  = (const float*)d_in[1];
    const float* tx  = (const float*)d_in[2];
    const float* ty  = (const float*)d_in[3];
    const float* ew  = (const float*)d_in[4];
    const float* eb  = (const float*)d_in[5];
    const float* qw  = (const float*)d_in[6];
    const float* qb  = (const float*)d_in[7];
    const float* kw  = (const float*)d_in[8];
    const float* kb  = (const float*)d_in[9];
    const float* vw  = (const float*)d_in[10];
    const float* vb  = (const float*)d_in[11];
    const float* ow  = (const float*)d_in[12];
    const float* ob  = (const float*)d_in[13];
    const float* n1s = (const float*)d_in[14];
    const float* n1b = (const float*)d_in[15];
    const float* n2s = (const float*)d_in[16];
    const float* n2b = (const float*)d_in[17];
    const float* f1w = (const float*)d_in[18];
    const float* f1b = (const float*)d_in[19];
    const float* f2w = (const float*)d_in[20];
    const float* f2b = (const float*)d_in[21];
    const float* mw  = (const float*)d_in[22];
    const float* mb  = (const float*)d_in[23];
    const float* sw  = (const float*)d_in[24];
    const float* sb  = (const float*)d_in[25];
    float* out = (float*)d_out;

    // bump allocator over workspace, 256B aligned
    char* p = (char*)d_ws;
    auto alloc = [&](size_t bytes) -> void* {
        void* r = (void*)p;
        p += (bytes + 255) & ~((size_t)255);
        return r;
    };
    float* Xin   = (float*)alloc((size_t)ROWS * XCAT * 4);
    float* stats = (float*)alloc(64 * 4);
    float* X     = (float*)alloc((size_t)ROWS * DMQ * 4);
    float* Y2    = (float*)alloc((size_t)ROWS * DMQ * 4);
    float* Qb    = (float*)alloc((size_t)ROWS * DMQ * 4);
    float* Kb    = (float*)alloc((size_t)ROWS * DMQ * 4);
    float* Vb    = (float*)alloc((size_t)ROWS * DMQ * 4);
    float* lparr = (float*)alloc(4096 * 4);
    float* sqarr = (float*)alloc(4096 * 4);
    _Float16* Xh   = (_Float16*)alloc((size_t)ROWS * DMQ * 2);
    _Float16* AOh  = (_Float16*)alloc((size_t)ROWS * DMQ * 2);
    _Float16* FFHh = (_Float16*)alloc((size_t)ROWS * DFFQ * 2);
    _Float16* QWt  = (_Float16*)alloc((size_t)6 * DMQ * DMQ * 2);
    _Float16* KWt  = (_Float16*)alloc((size_t)6 * DMQ * DMQ * 2);
    _Float16* VWt  = (_Float16*)alloc((size_t)6 * DMQ * DMQ * 2);
    _Float16* OWt  = (_Float16*)alloc((size_t)6 * DMQ * DMQ * 2);
    _Float16* F1Wt = (_Float16*)alloc((size_t)6 * DMQ * DFFQ * 2);
    _Float16* F2Wt = (_Float16*)alloc((size_t)6 * DFFQ * DMQ * 2);

    // --- weight transpose+cast (all layers) ---
    {
        int nW = DMQ * DMQ, gW = (nW + 255) / 256;
        int nF = DMQ * DFFQ, gFt = (nF + 255) / 256;
        for (int l = 0; l < 6; ++l) {
            transpose_cast_k<<<gW, 256, 0, stream>>>(qw + (size_t)l * nW, QWt + (size_t)l * nW, DMQ, DMQ);
            transpose_cast_k<<<gW, 256, 0, stream>>>(kw + (size_t)l * nW, KWt + (size_t)l * nW, DMQ, DMQ);
            transpose_cast_k<<<gW, 256, 0, stream>>>(vw + (size_t)l * nW, VWt + (size_t)l * nW, DMQ, DMQ);
            transpose_cast_k<<<gW, 256, 0, stream>>>(ow + (size_t)l * nW, OWt + (size_t)l * nW, DMQ, DMQ);
            transpose_cast_k<<<gFt, 256, 0, stream>>>(f1w + (size_t)l * nF, F1Wt + (size_t)l * nF, DMQ, DFFQ);
            transpose_cast_k<<<gFt, 256, 0, stream>>>(f2w + (size_t)l * nF, F2Wt + (size_t)l * nF, DFFQ, DMQ);
        }
    }

    // --- encode ---
    build_input_k<<<(ROWS * XCAT + 255) / 256, 256, 0, stream>>>(cx, cy, tx, Xin);
    bn_stats_k<<<1, 32, 0, stream>>>(Xin, stats);
    encode_k<<<(ROWS * DMQ + 255) / 256, 256, 0, stream>>>(Xin, stats, ew, eb, X, Xh);

    // macro-tiles: 32 rows x 64 cols per wave, 8 waves per block
    const int tilesD = (ROWS / 32) * (DMQ / 64);    // 256
    const int tilesF = (ROWS / 32) * (DFFQ / 64);   // 128
    const int gD = (tilesD + 7) / 8;
    const int gF = (tilesF + 7) / 8;

    for (int l = 0; l < 6; ++l) {
        const _Float16* qwt = QWt + (size_t)l * DMQ * DMQ;
        const _Float16* kwt = KWt + (size_t)l * DMQ * DMQ;
        const _Float16* vwt = VWt + (size_t)l * DMQ * DMQ;
        const _Float16* owt = OWt + (size_t)l * DMQ * DMQ;
        const _Float16* f1t = F1Wt + (size_t)l * DMQ * DFFQ;
        const _Float16* f2t = F2Wt + (size_t)l * DFFQ * DMQ;

        gemm_wmma_k<<<gD, 256, 0, stream>>>(Xh, qwt, qb + l * DMQ, Qb, nullptr,
                                            ROWS, DMQ, DMQ, 1);
        gemm_wmma_k<<<gD, 256, 0, stream>>>(Xh, kwt, kb + l * DMQ, Kb, nullptr,
                                            ROWS, DMQ, DMQ, 1);
        gemm_wmma_k<<<gD, 256, 0, stream>>>(Xh, vwt, vb + l * DMQ, Vb, nullptr,
                                            ROWS, DMQ, DMQ, 0);
        attn_k<<<BQ * HQ, 256, 0, stream>>>(Qb, Kb, Vb, AOh);
        gemm_wmma_k<<<gD, 256, 0, stream>>>(AOh, owt, ob + l * DMQ, Y2, nullptr,
                                            ROWS, DMQ, DMQ, 0);
        ln_res_k<<<ROWS, 128, 0, stream>>>(X, Xh, Y2, n1s + l * DMQ, n1b + l * DMQ);
        gemm_wmma_k<<<gF, 256, 0, stream>>>(Xh, f1t, f1b + l * DFFQ, nullptr, FFHh,
                                            ROWS, DFFQ, DMQ, 2);
        gemm_wmma_k<<<gD, 256, 0, stream>>>(FFHh, f2t, f2b + l * DMQ, Y2, nullptr,
                                            ROWS, DMQ, DFFQ, 0);
        ln_res_k<<<ROWS, 128, 0, stream>>>(X, Xh, Y2, n2s + l * DMQ, n2b + l * DMQ);
    }

    head_k<<<(BQ * TTQ * OUTQ + 255) / 256, 256, 0, stream>>>(X, ty, mw, mb, sw, sb,
                                                              out, lparr, sqarr);
    loss_k<<<1, 256, 0, stream>>>(lparr, sqarr, out);
}
